// Hypernet_58308476010883
// MI455X (gfx1250) — matrix operand order repaired and checked by reading, not compiled, not run on hardware
//
#include <hip/hip_runtime.h>

// ---------------------------------------------------------------------------
// Hypernet pipeline for MI455X (gfx1250), bf16x3 split-precision WMMA.
//   h1 = relu(x@W1+b1); h2 = relu(h1@W2+b2); h3 = relu(h2@W3+b3)
//   tail: per-sample w = h3@W4+b4 generated on the fly in LDS (never hits HBM)
// All matmuls use V_WMMA_F32_16X16X32_BF16 with hi/lo bf16 decomposition
// (3 WMMAs per product) for ~f32 accuracy at bf16 matrix-pipe speed.
// B matrices are pre-swizzled in global memory into 16-row slabs stored
// column-major-within-slab, so every WMMA fragment is loaded from LDS with
// b128 ops. The fused tail stages W4 tiles with GLOBAL_LOAD_ASYNC_TO_LDS_B128
// (ASYNCcnt path) so the copies bypass VGPRs and overlap with the WMMA stream.
// ---------------------------------------------------------------------------

typedef __attribute__((ext_vector_type(16))) __bf16    v16bf;
typedef __attribute__((ext_vector_type(8)))  float     v8f;
typedef __attribute__((ext_vector_type(4)))  unsigned  u32x4;

union BFrag {
    v16bf    v;
    unsigned w[8];
    u32x4    q[2];
};

enum { BSZ = 2048, IN1 = 1024, IN2 = 512, HF = 2048,
       BN = 100, BNP = 128, WSZ = 41024, HG = 64 };

__device__ __forceinline__ unsigned short f2bf(float f) {
    unsigned u = __float_as_uint(f);
    unsigned r = u + 0x7FFFu + ((u >> 16) & 1u);   // round-to-nearest-even
    return (unsigned short)(r >> 16);
}
__device__ __forceinline__ float bf2f(unsigned short h) {
    return __uint_as_float(((unsigned)h) << 16);
}

// ---- gfx1250 async global->LDS copy (16 bytes per lane), ASYNCcnt-tracked --
__device__ __forceinline__ void async_copy16(const unsigned short* gsrc,
                                             unsigned short* ldst) {
    unsigned lds_addr = (unsigned)(unsigned long long)(uintptr_t)ldst;
    unsigned long long gaddr = (unsigned long long)(uintptr_t)gsrc;
    asm volatile("global_load_async_to_lds_b128 %0, %1, off"
                 :
                 : "v"(lds_addr), "v"(gaddr)
                 : "memory");
}
__device__ __forceinline__ void wait_async0() {
    asm volatile("s_wait_asynccnt 0x0" ::: "memory");
}

// -------------------- f32 -> (bf16 hi, bf16 lo), flat (A matrices) ----------
__global__ void split_kernel(const float* __restrict__ in,
                             unsigned short* __restrict__ hi,
                             unsigned short* __restrict__ lo, long n) {
    long i = (long)blockIdx.x * blockDim.x + threadIdx.x;
    if (i < n) {
        float v = in[i];
        unsigned short h = f2bf(v);
        hi[i] = h;
        lo[i] = f2bf(v - bf2f(h));
    }
}

// ---- f32 [K,N] -> bf16 hi/lo in slab-swizzled B layout --------------------
// out[ ((k>>4)*N + n)*16 + (k&15) ]  for k in [0,Kpad), zero for k >= K.
// 2D grid: blockIdx.y = k row (no integer division).
__global__ void split_bsw_kernel(const float* __restrict__ in,
                                 unsigned short* __restrict__ hi,
                                 unsigned short* __restrict__ lo,
                                 int K, long N) {
    int  k = blockIdx.y;
    long n = (long)blockIdx.x * blockDim.x + threadIdx.x;
    if (n < N) {
        float v = (k < K) ? in[(long)k * N + n] : 0.0f;
        unsigned short h = f2bf(v);
        long o = (((long)(k >> 4)) * N + n) * 16 + (k & 15);
        hi[o] = h;
        lo[o] = f2bf(v - bf2f(h));
    }
}

__device__ __forceinline__ v8f wmma3(v8f acc, const BFrag& ah, const BFrag& al,
                                     const BFrag& bh, const BFrag& bl) {
    acc = __builtin_amdgcn_wmma_f32_16x16x32_bf16(false, al.v, false, bh.v,
                                                  (short)0, acc, false, false);
    acc = __builtin_amdgcn_wmma_f32_16x16x32_bf16(false, ah.v, false, bl.v,
                                                  (short)0, acc, false, false);
    acc = __builtin_amdgcn_wmma_f32_16x16x32_bf16(false, ah.v, false, bh.v,
                                                  (short)0, acc, false, false);
    return acc;
}

// ---------------------------------------------------------------------------
// bf16x3 GEMM:  C = act(A@B + bias).  A[M,K] row-major hi/lo bf16.
// B in slab-swizzled layout (units of 16 k-rows x 1 col = 32 bytes).
// Block 128 threads (4 waves); macro tile 128(M) x 64(N); K-chunk 32.
// Wave w computes rows [32w, 32w+32) x 64 cols  ->  8 accumulators.
// Output written as bf16 hi/lo, [M, Npad], cols in [Nb,Npad) = 0.
// ---------------------------------------------------------------------------
__launch_bounds__(128)
__global__ void gemm_bf16x3(const unsigned short* __restrict__ Ah,
                            const unsigned short* __restrict__ Al,
                            const unsigned short* __restrict__ Bh,   // swizzled
                            const unsigned short* __restrict__ Bl,   // swizzled
                            const float* __restrict__ bias,
                            unsigned short* __restrict__ Ch,
                            unsigned short* __restrict__ Cl,
                            int M, int K, int Nb, int Npad, int relu) {
    __shared__ unsigned short Ash[128][40], Asl[128][40];          // 16B-aligned rows
    __shared__ unsigned short Bswh[2 * 64 * 16], Bswl[2 * 64 * 16];

    const int tid  = threadIdx.x;
    const int wv   = tid >> 5;
    const int lane = tid & 31;
    const int lm   = lane & 15;
    const int kh   = lane >> 4;
    const int m0   = blockIdx.y * 128;
    const int n0   = blockIdx.x * 64;

    v8f zero = {};
    v8f acc[2][4];
#pragma unroll
    for (int mt = 0; mt < 2; ++mt)
#pragma unroll
        for (int nt = 0; nt < 4; ++nt) acc[mt][nt] = zero;

    for (int k0 = 0; k0 < K; k0 += 32) {
        // ---- stage A tile 128x32 (u32 copies) ----
#pragma unroll
        for (int i = 0; i < 16; ++i) {
            int e = tid + i * 128;
            int r = e >> 4, cp = e & 15;
            long idx = (long)(m0 + r) * K + k0 + cp * 2;
            *(unsigned*)&Ash[r][cp * 2] = *(const unsigned*)(Ah + idx);
            *(unsigned*)&Asl[r][cp * 2] = *(const unsigned*)(Al + idx);
        }
        // ---- stage B: 2 slabs x 64 cols, 32B unit copies ----
        {
            int u = tid;                  // 128 units, 1 per thread
            int slab = u >> 6, c = u & 63;
            int col  = n0 + c;
            u32x4* dh = (u32x4*)&Bswh[((slab << 6) + c) << 4];
            u32x4* dl = (u32x4*)&Bswl[((slab << 6) + c) << 4];
            if (col < Nb) {
                long kslab = (long)(k0 >> 4) + slab;
                const u32x4* sh = (const u32x4*)(Bh + (kslab * Nb + col) * 16);
                const u32x4* sl = (const u32x4*)(Bl + (kslab * Nb + col) * 16);
                dh[0] = sh[0]; dh[1] = sh[1];
                dl[0] = sl[0]; dl[1] = sl[1];
            } else {
                u32x4 z = {};
                dh[0] = z; dh[1] = z; dl[0] = z; dl[1] = z;
            }
        }
        __syncthreads();

        // ---- fragments: all b128 loads ----
        BFrag ah[2], al[2], bh[4], bl[4];
#pragma unroll
        for (int mt = 0; mt < 2; ++mt) {
            int row = (wv << 5) + (mt << 4) + lm;
            ah[mt].q[0] = *(const u32x4*)&Ash[row][16 * kh];
            ah[mt].q[1] = *(const u32x4*)&Ash[row][16 * kh + 8];
            al[mt].q[0] = *(const u32x4*)&Asl[row][16 * kh];
            al[mt].q[1] = *(const u32x4*)&Asl[row][16 * kh + 8];
        }
#pragma unroll
        for (int nt = 0; nt < 4; ++nt) {
            int base = (((kh << 6) + (nt << 4) + lm) << 4);
            bh[nt].q[0] = *(const u32x4*)&Bswh[base];
            bh[nt].q[1] = *(const u32x4*)&Bswh[base + 8];
            bl[nt].q[0] = *(const u32x4*)&Bswl[base];
            bl[nt].q[1] = *(const u32x4*)&Bswl[base + 8];
        }
#pragma unroll
        for (int mt = 0; mt < 2; ++mt)
#pragma unroll
            for (int nt = 0; nt < 4; ++nt)
                acc[mt][nt] = wmma3(acc[mt][nt], ah[mt], al[mt], bh[nt], bl[nt]);
        __syncthreads();
    }

    // ---- epilogue: bias + relu, re-split to bf16 hi/lo ----
#pragma unroll
    for (int mt = 0; mt < 2; ++mt)
#pragma unroll
        for (int nt = 0; nt < 4; ++nt) {
            int col = n0 + (nt << 4) + lm;
            if (col >= Npad) continue;
            float b = (col < Nb) ? bias[col] : 0.0f;
#pragma unroll
            for (int r = 0; r < 8; ++r) {
                int row = m0 + (wv << 5) + (mt << 4) + r + (kh << 3);
                float v = acc[mt][nt][r] + b;
                if (relu) v = fmaxf(v, 0.0f);
                unsigned short h = f2bf(v);
                long idx = (long)row * Npad + col;
                Ch[idx] = h;
                Cl[idx] = f2bf(v - bf2f(h));
            }
        }
}

// ---------------------------------------------------------------------------
// Fused tail: one block per 16 samples. Per 64-col chunk of (swizzled) W4:
//   wbuf[16][64] = h3[16x128] @ W4[:,chunk] + b4    (bf16x3 WMMA, K=128)
// then per-sample dots against y / previous activation (VALU, tiny).
// B tiles staged with global_load_async_to_lds_b128 (ASYNCcnt path).
// ---------------------------------------------------------------------------
__device__ __forceinline__ void chunk_gemm(
        const unsigned short* __restrict__ W4h,   // swizzled [8][WSZ][16]
        const unsigned short* __restrict__ W4l,
        const float* __restrict__ b4, long colbase,
        int tid, int wv, int lm, int kh,
        const BFrag* ahf, const BFrag* alf,
        unsigned short* Bsw_h, unsigned short* Bsw_l,
        float (*wbuf)[66]) {
    // stage B: 8 slabs x 64 cols, 32B units, async DMA into LDS
#pragma unroll
    for (int i = 0; i < 4; ++i) {
        int u = tid + i * 128;
        int slab = u >> 6, c = u & 63;
        long src = ((long)slab * WSZ + colbase + c) * 16;
        unsigned short* dh = &Bsw_h[((slab << 6) + c) << 4];
        unsigned short* dl = &Bsw_l[((slab << 6) + c) << 4];
        async_copy16(W4h + src,     dh);
        async_copy16(W4h + src + 8, dh + 8);
        async_copy16(W4l + src,     dl);
        async_copy16(W4l + src + 8, dl + 8);
        // prefetch next chunk of W4 into cache (global_prefetch_b8)
        if (colbase + 64 + c < WSZ)
            __builtin_prefetch(W4h + src + 64 * 16, 0, 1);
    }
    wait_async0();
    __syncthreads();

    v8f acc = {};
    int cc = (wv << 4) + lm;     // this wave's output column within [0,64)
#pragma unroll
    for (int ks = 0; ks < 4; ++ks) {
        BFrag bh, bl;
        int base = ((((ks * 2 + kh) << 6) + cc) << 4);
        bh.q[0] = *(const u32x4*)&Bsw_h[base];
        bh.q[1] = *(const u32x4*)&Bsw_h[base + 8];
        bl.q[0] = *(const u32x4*)&Bsw_l[base];
        bl.q[1] = *(const u32x4*)&Bsw_l[base + 8];
        acc = wmma3(acc, ahf[ks], alf[ks], bh, bl);
    }
    float b = b4[colbase + cc];
#pragma unroll
    for (int r = 0; r < 8; ++r)
        wbuf[r + (kh << 3)][cc] = acc[r] + b;
    __syncthreads();
}

__launch_bounds__(128)
__global__ void hyper_tail(const unsigned short* __restrict__ h3h,
                           const unsigned short* __restrict__ h3l,
                           const unsigned short* __restrict__ W4h,
                           const unsigned short* __restrict__ W4l,
                           const float* __restrict__ b4,
                           const float* __restrict__ y,
                           float* __restrict__ out) {
    __shared__ unsigned short Ah[16][136], Al[16][136];            // 16B-aligned rows
    __shared__ unsigned short Bsw_h[8 * 64 * 16], Bsw_l[8 * 64 * 16];
    __shared__ float wbuf[16][66];
    __shared__ float ych[16][64];
    __shared__ float outv[16][64];
    __shared__ float outn[16][64];
    __shared__ float red[16][8];

    const int tid  = threadIdx.x;
    const int wv   = tid >> 5;
    const int lane = tid & 31;
    const int lm   = lane & 15;
    const int kh   = lane >> 4;
    const int sp   = tid & 15;       // sample within group (dot phase)
    const int part = tid >> 4;       // partial-sum slot 0..7
    const int s0   = blockIdx.x * 16;

    // stage h3 (A matrix, 16 samples x 128 K, zero-padded cols >=100)
#pragma unroll
    for (int i = 0; i < 8; ++i) {
        int e = tid + i * 128;
        int r = e >> 6, cp = e & 63;
        long idx = (long)(s0 + r) * BNP + cp * 2;
        *(unsigned*)&Ah[r][cp * 2] = *(const unsigned*)(h3h + idx);
        *(unsigned*)&Al[r][cp * 2] = *(const unsigned*)(h3l + idx);
    }
    __syncthreads();

    // A fragments invariant across chunks: precompute all 4 K-steps (b128 loads)
    BFrag ahf[4], alf[4];
#pragma unroll
    for (int ks = 0; ks < 4; ++ks) {
        ahf[ks].q[0] = *(const u32x4*)&Ah[lm][ks * 32 + 16 * kh];
        ahf[ks].q[1] = *(const u32x4*)&Ah[lm][ks * 32 + 16 * kh + 8];
        alf[ks].q[0] = *(const u32x4*)&Al[lm][ks * 32 + 16 * kh];
        alf[ks].q[1] = *(const u32x4*)&Al[lm][ks * 32 + 16 * kh + 8];
    }

    // ---------------- layer 1: out[s,i] = relu( w[s, i*512+j] . y[s,j] ) ----
    for (int i = 0; i < HG; ++i) {
        float accp = 0.0f;
        for (int c = 0; c < 8; ++c) {
            __syncthreads();
            // stage y chunk [16][64]
#pragma unroll
            for (int i2 = 0; i2 < 8; ++i2) {
                int e = tid + i2 * 128;
                int r = e >> 6, cc = e & 63;
                ych[r][cc] = y[(long)(s0 + r) * IN2 + c * 64 + cc];
            }
            chunk_gemm(W4h, W4l, b4, (long)i * IN2 + c * 64,
                       tid, wv, lm, kh, ahf, alf, Bsw_h, Bsw_l, wbuf);
#pragma unroll
            for (int j = part * 8; j < part * 8 + 8; ++j)
                accp += wbuf[sp][j] * ych[sp][j];
        }
        red[sp][part] = accp;
        __syncthreads();
        if (tid < 16) {
            float v = 0.0f;
#pragma unroll
            for (int p = 0; p < 8; ++p) v += red[tid][p];
            outv[tid][i] = fmaxf(v, 0.0f);
        }
    }

    // ---------------- middle layers (no activation) -------------------------
    for (int t = 0; t < 2; ++t) {
        long base = (long)IN2 * HG + (long)t * HG * HG;
        for (int i = 0; i < HG; ++i) {
            __syncthreads();
            chunk_gemm(W4h, W4l, b4, base + (long)i * HG,
                       tid, wv, lm, kh, ahf, alf, Bsw_h, Bsw_l, wbuf);
            float accp = 0.0f;
#pragma unroll
            for (int j = part * 8; j < part * 8 + 8; ++j)
                accp += wbuf[sp][j] * outv[sp][j];
            red[sp][part] = accp;
            __syncthreads();
            if (tid < 16) {
                float v = 0.0f;
#pragma unroll
                for (int p = 0; p < 8; ++p) v += red[tid][p];
                outn[tid][i] = v;
            }
        }
        __syncthreads();
        for (int e = tid; e < 16 * 64; e += 128)
            (&outv[0][0])[e] = (&outn[0][0])[e];
        __syncthreads();
    }

    // ---------------- output layer ------------------------------------------
    {
        __syncthreads();
        chunk_gemm(W4h, W4l, b4, (long)IN2 * HG + 2L * HG * HG,
                   tid, wv, lm, kh, ahf, alf, Bsw_h, Bsw_l, wbuf);
        float accp = 0.0f;
#pragma unroll
        for (int j = part * 8; j < part * 8 + 8; ++j)
            accp += wbuf[sp][j] * outv[sp][j];
        red[sp][part] = accp;
        __syncthreads();
        if (tid < 16) {
            float v = 0.0f;
#pragma unroll
            for (int p = 0; p < 8; ++p) v += red[tid][p];
            out[s0 + tid] = v;
        }
    }
}

// ---------------------------------------------------------------------------
extern "C" void kernel_launch(void* const* d_in, const int* in_sizes, int n_in,
                              void* d_out, int out_size, void* d_ws, size_t ws_size,
                              hipStream_t stream) {
    (void)in_sizes; (void)n_in; (void)out_size; (void)ws_size;

    const float* x  = (const float*)d_in[0];
    const float* y  = (const float*)d_in[1];
    const float* W1 = (const float*)d_in[2];
    const float* b1 = (const float*)d_in[3];
    const float* W2 = (const float*)d_in[4];
    const float* b2 = (const float*)d_in[5];
    const float* W3 = (const float*)d_in[6];
    const float* b3 = (const float*)d_in[7];
    const float* W4 = (const float*)d_in[8];
    const float* b4 = (const float*)d_in[9];
    float* out = (float*)d_out;

    size_t off = 0;
    auto take = [&](size_t elems) -> unsigned short* {
        unsigned short* r = (unsigned short*)((char*)d_ws + off);
        off += (elems * sizeof(unsigned short) + 255) & ~(size_t)255;
        return r;
    };
    unsigned short* xh  = take((size_t)BSZ * IN1);
    unsigned short* xl  = take((size_t)BSZ * IN1);
    unsigned short* w1h = take((size_t)IN1 * HF);
    unsigned short* w1l = take((size_t)IN1 * HF);
    unsigned short* h1h = take((size_t)BSZ * HF);
    unsigned short* h1l = take((size_t)BSZ * HF);
    unsigned short* w2h = take((size_t)HF * HF);
    unsigned short* w2l = take((size_t)HF * HF);
    unsigned short* h2h = take((size_t)BSZ * HF);
    unsigned short* h2l = take((size_t)BSZ * HF);
    unsigned short* w3h = take((size_t)HF * BN);
    unsigned short* w3l = take((size_t)HF * BN);
    unsigned short* h3h = take((size_t)BSZ * BNP);
    unsigned short* h3l = take((size_t)BSZ * BNP);
    unsigned short* w4h = take((size_t)BNP * WSZ);
    unsigned short* w4l = take((size_t)BNP * WSZ);

    const int thr = 256;
    // A matrix of GEMM1: plain row-major split
    {
        long n = (long)BSZ * IN1;
        split_kernel<<<(unsigned)((n + thr - 1) / thr), thr, 0, stream>>>(x, xh, xl, n);
    }
    // B matrices: slab-swizzled split, 2D grid (y = k row; no int division)
    auto splitB = [&](const float* in, unsigned short* h, unsigned short* l,
                      int K, int Kpad, long N) {
        dim3 g((unsigned)((N + thr - 1) / thr), (unsigned)Kpad);
        split_bsw_kernel<<<g, thr, 0, stream>>>(in, h, l, K, N);
    };
    splitB(W1, w1h, w1l, IN1, IN1, HF);
    splitB(W2, w2h, w2l, HF,  HF,  HF);
    splitB(W3, w3h, w3l, HF,  HF,  BN);
    splitB(W4, w4h, w4l, BN,  BNP, WSZ);

    dim3 blk(128);
    // h1 = relu(x @ W1 + b1)
    gemm_bf16x3<<<dim3(HF / 64, BSZ / 128), blk, 0, stream>>>(
        xh, xl, w1h, w1l, b1, h1h, h1l, BSZ, IN1, HF, HF, 1);
    // h2 = relu(h1 @ W2 + b2)
    gemm_bf16x3<<<dim3(HF / 64, BSZ / 128), blk, 0, stream>>>(
        h1h, h1l, w2h, w2l, b2, h2h, h2l, BSZ, HF, HF, HF, 1);
    // h3 = relu(h2 @ W3 + b3), padded to 128 cols (zeros beyond 100)
    gemm_bf16x3<<<dim3(BNP / 64, BSZ / 128), blk, 0, stream>>>(
        h2h, h2l, w3h, w3l, b3, h3h, h3l, BSZ, HF, BN, BNP, 1);
    // fused w-generation + target net
    hyper_tail<<<dim3(BSZ / 16), blk, 0, stream>>>(h3h, h3l, w4h, w4l, b4, y, out);
}
